// Bigram_37529424233126
// MI455X (gfx1250) — matrix operand order repaired and checked by reading, not compiled, not run on hardware
//
#include <hip/hip_runtime.h>
#include <hip/hip_bf16.h>

// ---------------------------------------------------------------------------
// Bigram transformer forward on gfx1250 (MI455X).
// All GEMM operands pre-packed into WMMA lane layout (bf16) so the hot loops
// are pure global_load_b128 + v_wmma_f32_16x16x32_bf16. Flash attention with
// LDS transpose for the P operand. Memory-bound on the 1.05GB logits write.
// ---------------------------------------------------------------------------

typedef __bf16 bf16;
typedef bf16  v16bf __attribute__((ext_vector_type(16)));
typedef float v8f   __attribute__((ext_vector_type(8)));

#define BATCH  4
#define TSEQ   2048
#define EMB    256
#define NHEAD  4
#define HS     64
#define NLAYER 3
#define VOCAB  32000
#define MROWS  (BATCH * TSEQ)   // 8192
#define TTILES (TSEQ / 16)      // 128

// ---- WMMA operand element placement (CDNA5 ISA 7.12.2, wave32) -------------
// A (16x32, 16-bit): lane L -> m = L&15, s = L>>4.
//   VGPR v in 0..3 : K = 2v + 8s + hi ; v in 4..7 : K = 16 + 2(v-4) + 8s + hi
__device__ __forceinline__ int a_k(int e, int s) {
  int v = e >> 1, hi = e & 1;
  int base = (v < 4) ? (2 * v) : (16 + 2 * (v - 4));
  return base + 8 * s + hi;
}
// B (32x16, 16-bit): lane L -> n = L&15, s = L>>4; same K order as A
__device__ __forceinline__ int b_k(int e, int s) {
  int v = e >> 1, hi = e & 1;
  int base = (v < 4) ? (2 * v) : (16 + 2 * (v - 4));
  return base + 8 * s + hi;
}
// C/D (16x16 f32): lane L -> n = L&15; vgpr r holds row m = r + 8*(L>>4)

// ---------------------------------------------------------------------------
// Embedding: x[b*T+t, :] = tok_emb[idx[b*T+t], :] + pos_emb[t, :]
// ---------------------------------------------------------------------------
__global__ __launch_bounds__(256) void embed_kernel(
    const int* __restrict__ idx, const float* __restrict__ tok,
    const float* __restrict__ pos, float* __restrict__ x) {
  int i = blockIdx.x;
  int e = threadIdx.x;
  int t = i & (TSEQ - 1);
  x[(long)i * EMB + e] = tok[(long)idx[i] * EMB + e] + pos[(long)t * EMB + e];
}

// ---------------------------------------------------------------------------
// LayerNorm over EMB=256, bf16 output (row-major)
// ---------------------------------------------------------------------------
__global__ __launch_bounds__(256) void ln_kernel(
    const float* __restrict__ x, const float* __restrict__ g,
    const float* __restrict__ bta, bf16* __restrict__ out) {
  __shared__ float red[256];
  int row = blockIdx.x, tid = threadIdx.x;
  float v = x[(long)row * EMB + tid];
  red[tid] = v;
  __syncthreads();
  for (int s = 128; s > 0; s >>= 1) {
    if (tid < s) red[tid] += red[tid + s];
    __syncthreads();
  }
  float mean = red[0] * (1.0f / EMB);
  __syncthreads();
  float d = v - mean;
  red[tid] = d * d;
  __syncthreads();
  for (int s = 128; s > 0; s >>= 1) {
    if (tid < s) red[tid] += red[tid + s];
    __syncthreads();
  }
  float var = red[0] * (1.0f / EMB);
  out[(long)row * EMB + tid] =
      (bf16)(d * rsqrtf(var + 1e-5f) * g[tid] + bta[tid]);
}

// fp32 -> bf16 row copy
__global__ __launch_bounds__(256) void cvt_kernel(const float* __restrict__ x,
                                                  bf16* __restrict__ xb) {
  long i = (long)blockIdx.x * 256 + threadIdx.x;
  xb[i] = (bf16)x[i];
}

// ---------------------------------------------------------------------------
// Operand packers: one thread per packed element, writes fully coalesced.
// Packed block = 512 bf16 (32 lanes x 16 elements) = one wave WMMA operand.
// ---------------------------------------------------------------------------
// fp32 row-major B[K,N] -> bf16 B-operand tiles [(K/32)*(N/16)][512]
__global__ __launch_bounds__(256) void pack_b_kernel(
    const float* __restrict__ B, bf16* __restrict__ Bp, int K, int N) {
  long tid = (long)blockIdx.x * 256 + threadIdx.x;
  int e = (int)(tid & 15);
  int lane = (int)((tid >> 4) & 31);
  long blk = tid >> 9;
  int nTiles = N >> 4;
  int nc = (int)(blk % nTiles);
  int kb = (int)(blk / nTiles);
  int sh = lane >> 4, lm = lane & 15;
  Bp[tid] = (bf16)B[(long)(kb * 32 + b_k(e, sh)) * N + nc * 16 + lm];
}

// bf16 row-major A[M,K] -> A-operand tiles [(M/16)*(K/32)][512]
__global__ __launch_bounds__(256) void pack_a_kernel(
    const bf16* __restrict__ A, bf16* __restrict__ Ap, int K) {
  long tid = (long)blockIdx.x * 256 + threadIdx.x;
  int e = (int)(tid & 15);
  int lane = (int)((tid >> 4) & 31);
  long blk = tid >> 9;
  int kTiles = K >> 5;
  int kb = (int)(blk % kTiles);
  long mt = blk / kTiles;
  Ap[tid] = A[(mt * 16 + (lane & 15)) * (long)K + kb * 32 + a_k(e, lane >> 4)];
}

// q -> A-layout per (bh, t-tile, d-chunk):  [bh][TTILES][2][512]
__global__ __launch_bounds__(256) void pack_q_kernel(
    const bf16* __restrict__ q, bf16* __restrict__ qp) {
  long tid = (long)blockIdx.x * 256 + threadIdx.x;
  int e = (int)(tid & 15);
  int lane = (int)((tid >> 4) & 31);
  long blk = tid >> 9;
  int ch = (int)(blk & 1);
  int jm = (int)((blk >> 1) & (TTILES - 1));
  int bh = (int)(blk >> 8);
  long rowbase = (long)(bh >> 2) * TSEQ;
  int colbase = (bh & 3) * HS;
  int sh = lane >> 4, lm = lane & 15;
  qp[tid] = q[(rowbase + jm * 16 + lm) * EMB + colbase + ch * 32 + a_k(e, sh)];
}

// k -> B-layout (K = head dim, N = key time):  [bh][TTILES][2][512]
__global__ __launch_bounds__(256) void pack_k_kernel(
    const bf16* __restrict__ k, bf16* __restrict__ kp) {
  long tid = (long)blockIdx.x * 256 + threadIdx.x;
  int e = (int)(tid & 15);
  int lane = (int)((tid >> 4) & 31);
  long blk = tid >> 9;
  int ch = (int)(blk & 1);
  int jm = (int)((blk >> 1) & (TTILES - 1));
  int bh = (int)(blk >> 8);
  long rowbase = (long)(bh >> 2) * TSEQ;
  int colbase = (bh & 3) * HS;
  int sh = lane >> 4, lm = lane & 15;
  kp[tid] = k[(rowbase + jm * 16 + lm) * EMB + colbase + ch * 32 + b_k(e, sh)];
}

// v -> B-layout (K = key time, zero-padded 16..31; N = head dim chunks):
//   [bh][TTILES][4][512]
__global__ __launch_bounds__(256) void pack_v_kernel(
    const bf16* __restrict__ v, bf16* __restrict__ vp) {
  long tid = (long)blockIdx.x * 256 + threadIdx.x;
  int e = (int)(tid & 15);
  int lane = (int)((tid >> 4) & 31);
  long blk = tid >> 9;
  int c = (int)(blk & 3);
  int jm = (int)((blk >> 2) & (TTILES - 1));
  int bh = (int)(blk >> 9);
  long rowbase = (long)(bh >> 2) * TSEQ;
  int colbase = (bh & 3) * HS;
  int sh = lane >> 4, lm = lane & 15;
  int kk = b_k(e, sh);
  vp[tid] = (kk < 16)
                ? v[(rowbase + jm * 16 + kk) * EMB + colbase + c * 16 + lm]
                : (bf16)0.0f;
}

// ---------------------------------------------------------------------------
// GEMM on pre-packed operands: one wave per 16x64 tile.
// Inner loop: 2xb128 (A) + 8xb128 (B) + 4 WMMA per K-step.
// ---------------------------------------------------------------------------
__global__ __launch_bounds__(32) void gemm_kernel(
    const v16bf* __restrict__ Ap, const v16bf* __restrict__ Bp,
    const float* __restrict__ bias, const float* __restrict__ resid,
    float* __restrict__ outF, bf16* __restrict__ outB, int K, int N,
    int relu) {
  int lane = threadIdx.x & 31;
  int lm = lane & 15, sh = lane >> 4;
  int mt = blockIdx.x;
  int nc4 = blockIdx.y;
  int K32 = K >> 5, N16 = N >> 4;
  v8f acc[4] = {};
  const v16bf* ap = Ap + (long)mt * K32 * 32 + lane;
  for (int kt = 0; kt < K32; kt++) {
    v16bf a = ap[(long)kt * 32];
    if (kt + 1 < K32)
      __builtin_prefetch((const void*)(ap + (long)(kt + 1) * 32), 0, 0);
    const v16bf* bp = Bp + ((long)kt * N16 + nc4 * 4) * 32 + lane;
#pragma unroll
    for (int c = 0; c < 4; c++) {
      v16bf bm = bp[c * 32];
      acc[c] = __builtin_amdgcn_wmma_f32_16x16x32_bf16(
          false, a, false, bm, (short)0, acc[c], false, false);
    }
  }
#pragma unroll
  for (int c = 0; c < 4; c++) {
#pragma unroll
    for (int r = 0; r < 8; r++) {
      long row = (long)mt * 16 + r + 8 * sh;
      int col = nc4 * 64 + c * 16 + lm;
      float val = acc[c][r];
      if (bias) val += bias[col];
      if (resid) val += resid[row * (long)N + col];
      if (relu) val = fmaxf(val, 0.0f);
      long off = row * (long)N + col;
      if (outF) outF[off] = val;
      else outB[off] = (bf16)val;
    }
  }
}

// ---------------------------------------------------------------------------
// Flash attention on packed q/k/v, causal, scale = 1/sqrt(EMB) = 1/16.
// One wave per (16 query rows, batch*head). LDS transposes P (C->A layout).
// ---------------------------------------------------------------------------
__global__ __launch_bounds__(32) void attn_kernel(
    const v16bf* __restrict__ qp, const v16bf* __restrict__ kp,
    const v16bf* __restrict__ vp, bf16* __restrict__ y) {
  const float scale = 0.0625f;  // 1/sqrt(256), matches reference
  int lane = threadIdx.x & 31;
  int lm = lane & 15, sh = lane >> 4;
  int mt = blockIdx.x;
  int bh = blockIdx.y;
  long rowbase = (long)(bh >> 2) * TSEQ;
  int colbase = (bh & 3) * HS;

  v16bf qa[2];
#pragma unroll
  for (int ch = 0; ch < 2; ch++)
    qa[ch] = qp[((long)(bh * TTILES + mt) * 2 + ch) * 32 + lane];

  float mrow[8], lrow[8];
  v8f o[4] = {};
#pragma unroll
  for (int r = 0; r < 8; r++) { mrow[r] = -1e30f; lrow[r] = 0.0f; }

  __shared__ float Pst[16][17];

  for (int j = 0; j <= mt; j++) {
    v8f sacc = {};
#pragma unroll
    for (int ch = 0; ch < 2; ch++) {
      v16bf kb = kp[((long)(bh * TTILES + j) * 2 + ch) * 32 + lane];
      sacc = __builtin_amdgcn_wmma_f32_16x16x32_bf16(
          false, qa[ch], false, kb, (short)0, sacc, false, false);
    }
    float p[8];
#pragma unroll
    for (int r = 0; r < 8; r++) {
      float sv = sacc[r] * scale;
      if (j == mt && (j * 16 + lm) > (mt * 16 + r + 8 * sh)) sv = -1e30f;
      p[r] = sv;
    }
#pragma unroll
    for (int r = 0; r < 8; r++) {
      float xm = p[r];
#pragma unroll
      for (int off = 1; off < 16; off <<= 1)
        xm = fmaxf(xm, __shfl_xor(xm, off, 32));
      float mnew = fmaxf(mrow[r], xm);
      float pe = __expf(p[r] - mnew);
      float alpha = __expf(mrow[r] - mnew);
      float rs = pe;
#pragma unroll
      for (int off = 1; off < 16; off <<= 1) rs += __shfl_xor(rs, off, 32);
      lrow[r] = lrow[r] * alpha + rs;
      mrow[r] = mnew;
      p[r] = pe;
#pragma unroll
      for (int c = 0; c < 4; c++) o[c][r] *= alpha;
    }
    __syncthreads();
#pragma unroll
    for (int r = 0; r < 8; r++) Pst[r + 8 * sh][lm] = p[r];
    __syncthreads();
    v16bf pa;
#pragma unroll
    for (int e = 0; e < 16; e++) {
      int kk = a_k(e, sh);
      pa[e] = (kk < 16) ? (bf16)Pst[lm][kk] : (bf16)0.0f;
    }
#pragma unroll
    for (int c = 0; c < 4; c++) {
      v16bf vb = vp[((long)(bh * TTILES + j) * 4 + c) * 32 + lane];
      o[c] = __builtin_amdgcn_wmma_f32_16x16x32_bf16(
          false, pa, false, vb, (short)0, o[c], false, false);
    }
  }
#pragma unroll
  for (int c = 0; c < 4; c++)
#pragma unroll
    for (int r = 0; r < 8; r++)
      y[(rowbase + mt * 16 + r + 8 * sh) * EMB + colbase + c * 16 + lm] =
          (bf16)(o[c][r] / lrow[r]);
}

// ---------------------------------------------------------------------------
extern "C" void kernel_launch(void* const* d_in, const int* in_sizes, int n_in,
                              void* d_out, int out_size, void* d_ws,
                              size_t ws_size, hipStream_t stream) {
  (void)in_sizes; (void)n_in; (void)out_size; (void)ws_size;
  const int*   idx     = (const int*)d_in[0];
  const float* tok_emb = (const float*)d_in[1];
  const float* pos_emb = (const float*)d_in[2];
  const float* Wq      = (const float*)d_in[3];
  const float* Wk      = (const float*)d_in[4];
  const float* Wv      = (const float*)d_in[5];
  const float* Wproj   = (const float*)d_in[6];
  const float* bproj   = (const float*)d_in[7];
  const float* ln1_g   = (const float*)d_in[8];
  const float* ln1_b   = (const float*)d_in[9];
  const float* W1      = (const float*)d_in[10];
  const float* b1      = (const float*)d_in[11];
  const float* W2      = (const float*)d_in[12];
  const float* b2      = (const float*)d_in[13];
  const float* ln2_g   = (const float*)d_in[14];
  const float* ln2_b   = (const float*)d_in[15];
  const float* Wlm     = (const float*)d_in[16];
  const float* blm     = (const float*)d_in[17];
  float* out = (float*)d_out;

  // ---- workspace carve-up (~88 MB, all blocks 1KB-aligned sizes) ----
  char* w = (char*)d_ws;
  float* x     = (float*)w; w += (size_t)MROWS * EMB * 4;          // 8 MB
  bf16* bufRow = (bf16*)w;  w += (size_t)MROWS * 4 * EMB * 2;      // 16.8 MB (h/y/mid/xb)
  bf16* qrow   = (bf16*)w;  w += (size_t)MROWS * EMB * 2;          // 4.2 MB
  bf16* krow   = (bf16*)w;  w += (size_t)MROWS * EMB * 2;
  bf16* vrow   = (bf16*)w;  w += (size_t)MROWS * EMB * 2;
  bf16* qpk    = (bf16*)w;  w += (size_t)16 * TTILES * 2 * 512 * 2; // 4.2 MB
  bf16* kpk    = (bf16*)w;  w += (size_t)16 * TTILES * 2 * 512 * 2;
  bf16* vpk    = (bf16*)w;  w += (size_t)16 * TTILES * 4 * 512 * 2; // 8.4 MB
  bf16* apack  = (bf16*)w;  w += (size_t)(MROWS / 16) * (1024 / 32) * 512 * 2; // 16.8 MB
  bf16* wpack  = (bf16*)w;  w += (size_t)(256 / 32) * (VOCAB / 16) * 512 * 2;  // 16.4 MB

  dim3 blkWave(32);
  dim3 grid256(256);
  dim3 gridE(MROWS / 16, EMB / 64);
  dim3 gridM1(MROWS / 16, (4 * EMB) / 64);
  dim3 gridLM(MROWS / 16, VOCAB / 64);
  dim3 gridAttn(TTILES, BATCH * NHEAD);

  // pack grid sizes (256 threads each, one per packed element)
  const int gPackA_E  = (MROWS / 16) * (EMB / 32) * 2;        // K=256 activations
  const int gPackA_M  = (MROWS / 16) * ((4 * EMB) / 32) * 2;  // K=1024 (mid)
  const int gPackB_EE = (EMB / 32) * (EMB / 16) * 2;
  const int gPackB_E4 = (EMB / 32) * ((4 * EMB) / 16) * 2;
  const int gPackB_4E = ((4 * EMB) / 32) * (EMB / 16) * 2;
  const int gPackB_LM = (EMB / 32) * (VOCAB / 16) * 2;
  const int gPackQ    = 16 * TTILES * 2 * 2;
  const int gPackV    = 16 * TTILES * 4 * 2;

  embed_kernel<<<MROWS, 256, 0, stream>>>(idx, tok_emb, pos_emb, x);

  for (int l = 0; l < NLAYER; l++) {
    const float* wq = Wq + (size_t)l * EMB * EMB;
    const float* wk = Wk + (size_t)l * EMB * EMB;
    const float* wv = Wv + (size_t)l * EMB * EMB;
    const float* wp = Wproj + (size_t)l * EMB * EMB;
    const float* w1 = W1 + (size_t)l * EMB * 4 * EMB;
    const float* w2 = W2 + (size_t)l * 4 * EMB * EMB;

    // h = LN1(x); pack as A operand
    ln_kernel<<<MROWS, 256, 0, stream>>>(x, ln1_g + l * EMB, ln1_b + l * EMB,
                                         bufRow);
    pack_a_kernel<<<gPackA_E, 256, 0, stream>>>(bufRow, apack, EMB);

    // q,k,v = h @ Wq/Wk/Wv
    pack_b_kernel<<<gPackB_EE, 256, 0, stream>>>(wq, wpack, EMB, EMB);
    gemm_kernel<<<gridE, blkWave, 0, stream>>>((v16bf*)apack, (v16bf*)wpack,
        nullptr, nullptr, nullptr, qrow, EMB, EMB, 0);
    pack_b_kernel<<<gPackB_EE, 256, 0, stream>>>(wk, wpack, EMB, EMB);
    gemm_kernel<<<gridE, blkWave, 0, stream>>>((v16bf*)apack, (v16bf*)wpack,
        nullptr, nullptr, nullptr, krow, EMB, EMB, 0);
    pack_b_kernel<<<gPackB_EE, 256, 0, stream>>>(wv, wpack, EMB, EMB);
    gemm_kernel<<<gridE, blkWave, 0, stream>>>((v16bf*)apack, (v16bf*)wpack,
        nullptr, nullptr, nullptr, vrow, EMB, EMB, 0);

    // pack per-head WMMA operands, run flash attention -> y (bufRow)
    pack_q_kernel<<<gPackQ, 256, 0, stream>>>(qrow, qpk);
    pack_k_kernel<<<gPackQ, 256, 0, stream>>>(krow, kpk);
    pack_v_kernel<<<gPackV, 256, 0, stream>>>(vrow, vpk);
    attn_kernel<<<gridAttn, blkWave, 0, stream>>>((v16bf*)qpk, (v16bf*)kpk,
                                                  (v16bf*)vpk, bufRow);

    // x += y @ Wproj + bproj
    pack_a_kernel<<<gPackA_E, 256, 0, stream>>>(bufRow, apack, EMB);
    pack_b_kernel<<<gPackB_EE, 256, 0, stream>>>(wp, wpack, EMB, EMB);
    gemm_kernel<<<gridE, blkWave, 0, stream>>>((v16bf*)apack, (v16bf*)wpack,
        bproj + l * EMB, x, x, nullptr, EMB, EMB, 0);

    // h2 = LN2(x);  mid = relu(h2 @ W1 + b1);  x += mid @ W2 + b2
    ln_kernel<<<MROWS, 256, 0, stream>>>(x, ln2_g + l * EMB, ln2_b + l * EMB,
                                         bufRow);
    pack_a_kernel<<<gPackA_E, 256, 0, stream>>>(bufRow, apack, EMB);
    pack_b_kernel<<<gPackB_E4, 256, 0, stream>>>(w1, wpack, EMB, 4 * EMB);
    gemm_kernel<<<gridM1, blkWave, 0, stream>>>((v16bf*)apack, (v16bf*)wpack,
        b1 + (size_t)l * 4 * EMB, nullptr, nullptr, bufRow, EMB, 4 * EMB, 1);
    pack_a_kernel<<<gPackA_M, 256, 0, stream>>>(bufRow, apack, 4 * EMB);
    pack_b_kernel<<<gPackB_4E, 256, 0, stream>>>(w2, wpack, 4 * EMB, EMB);
    gemm_kernel<<<gridE, blkWave, 0, stream>>>((v16bf*)apack, (v16bf*)wpack,
        b2 + l * EMB, x, x, nullptr, 4 * EMB, EMB, 0);
  }

  // logits = x @ Wlm + blm
  cvt_kernel<<<MROWS, 256, 0, stream>>>(x, bufRow);
  pack_a_kernel<<<gPackA_E, 256, 0, stream>>>(bufRow, apack, EMB);
  pack_b_kernel<<<gPackB_LM, 256, 0, stream>>>(Wlm, wpack, EMB, VOCAB);
  gemm_kernel<<<gridLM, blkWave, 0, stream>>>((v16bf*)apack, (v16bf*)wpack,
      blm, nullptr, out, nullptr, EMB, VOCAB, 0);
  (void)grid256;
}
